// AttenPool_22299470201469
// MI455X (gfx1250) — matrix operand order repaired and checked by reading, not compiled, not run on hardware
//
#include <hip/hip_runtime.h>
#include <stdint.h>

#define G   512
#define S   128
#define DIN 256
#define H   4
#define C   64
#define HC  256

// padded LDS strides (elements): per-lane row stride = odd multiple of 16B -> conflict-free
#define LDQ    72     // Qh/Kh row stride (144B)
#define LDV    136    // Vt row stride (272B)
#define LDSLAB 136    // attn slab row stride (272B)
#define LDB    264    // staged weight-chunk row stride (528B)
#define STGE   (64 * LDB)   // one stage buffer: 64 rows

typedef __attribute__((ext_vector_type(16))) _Float16 v16h;
typedef __attribute__((ext_vector_type(8)))  float    v8f;

// ---------------- prep: fp32 weights [DIN][HC] -> f16 transposed [HC][DIN] ---
__global__ __launch_bounds__(256) void prep_weights(
    const float* __restrict__ Wq, const float* __restrict__ Wk,
    const float* __restrict__ Wv, const float* __restrict__ Ws,
    _Float16* __restrict__ WT)
{
    int idx = blockIdx.x * blockDim.x + threadIdx.x;   // 0 .. 4*65536-1
    int p   = idx >> 16;
    int rem = idx & 65535;
    int n   = rem >> 8;
    int k   = rem & 255;
    const float* W = (p == 0) ? Wq : (p == 1) ? Wk : (p == 2) ? Wv : Ws;
    WT[idx] = (_Float16)W[k * HC + n];                 // WT[p][n][k]
}

// ---------------- prep: x fp32 -> f16 (A operands stream from L2/HBM) --------
__global__ __launch_bounds__(256) void prep_x(
    const float* __restrict__ x, _Float16* __restrict__ x16)
{
    int i = (blockIdx.x * blockDim.x + threadIdx.x) * 4;
    float4 f = *(const float4*)(x + i);
    x16[i + 0] = (_Float16)f.x;
    x16[i + 1] = (_Float16)f.y;
    x16[i + 2] = (_Float16)f.z;
    x16[i + 3] = (_Float16)f.w;
}

// ---------------- fragment loaders (CDNA5 wave32 WMMA layouts) ---------------
// A (16x32 f16): lane l holds row m=l&15; k = (l>=16?16:0) + j
__device__ __forceinline__ v16h frag_A_rowmajor(const _Float16* base, int ld,
                                                int row0, int k0, int lane)
{
    const _Float16* p = base + (row0 + (lane & 15)) * ld + k0 + ((lane >> 4) << 4);
    union { v16h v; float4 f[2]; } u;
    u.f[0] = *(const float4*)(p);        // k = base+0..7
    u.f[1] = *(const float4*)(p + 8);    // k = base+8..15
    return u.v;
}
// B (32x16 f16) from an n-major source (row = n, col = k), contiguous in k:
// lane l holds col n=l&15; k = (l>=16?16:0) + 0..15
__device__ __forceinline__ v16h frag_B_nmajor(const _Float16* base, int ld,
                                              int n0, int k0, int lane)
{
    const _Float16* p = base + (n0 + (lane & 15)) * ld + k0 + ((lane >> 4) << 4);
    union { v16h v; float4 f[2]; } u;
    u.f[0] = *(const float4*)(p);
    u.f[1] = *(const float4*)(p + 8);
    return u.v;
}

// ---------------- async stage: 64 rows x 512B -> LDS (stride LDB) ------------
// 8 waves x 8 rows; each instruction moves 16B/lane x 32 lanes = one 512B row.
__device__ __forceinline__ void stage_chunk(_Float16* dst /*LDS*/,
                                            const _Float16* src /*global*/,
                                            int w, int lane)
{
#pragma unroll
    for (int rr = 0; rr < 8; ++rr) {
        const int row = w * 8 + rr;
        unsigned lds_off =
            (unsigned)(uintptr_t)(dst + row * LDB) + (unsigned)(lane * 16);
        unsigned long long ga =
            (unsigned long long)(uintptr_t)(src + row * DIN) + (unsigned)(lane * 16);
        asm volatile("global_load_async_to_lds_b128 %0, %1, off"
                     :: "v"(lds_off), "v"(ga) : "memory");
    }
}

// ---------------- fused attention + skip + max-pool, one block per group -----
__global__ __launch_bounds__(256) void atten_pool_kernel(
    const _Float16* __restrict__ x16,    // [N][DIN] f16
    const _Float16* __restrict__ WT,     // [4][HC][DIN] f16, n-major
    const float* __restrict__ bq, const float* __restrict__ bk,
    const float* __restrict__ bv, const float* __restrict__ bs,
    float* __restrict__ out)
{
    __shared__ _Float16 Qh[S * LDQ];            // 18 KB
    __shared__ _Float16 Kh[S * LDQ];            // 18 KB
    __shared__ _Float16 Vt[C * LDV];            // 17 KB (V transposed: [c][j])
    __shared__ _Float16 slab[8 * 16 * LDSLAB];  // 34 KB (per-wave attn rows)
    __shared__ _Float16 stage[2 * STGE];        // 66 KB (double-buffered W chunk)
    __shared__ float    pool[8 * C];            //  2 KB   => ~155 KB total

    const int g    = blockIdx.x;
    const int tid  = threadIdx.x;
    const int lane = tid & 31;
    const int w    = tid >> 5;        // wave id 0..7, owns rows [16w, 16w+16)
    const int half = lane >> 4;
    const int m0   = w * 16;
    const float scale = 0.125f;       // 1/sqrt(C)

    const _Float16* xga = x16 + (size_t)g * S * DIN;   // A operand base (global)

    for (int h = 0; h < H; ++h) {
        // prologue: stage mat 0 chunk (rows = cols h*C..h*C+63 of WT[0])
        stage_chunk(stage, WT + (size_t)(0 * HC + h * C) * DIN, w, lane);

        v8f skipAcc[4];
        for (int mat = 0; mat < 4; ++mat) {
            if (mat < 3) {
                stage_chunk(stage + ((mat + 1) & 1) * STGE,
                            WT + ((size_t)(mat + 1) * HC + h * C) * DIN, w, lane);
                asm volatile("s_wait_asynccnt 0x8" ::: "memory");
            } else {
                asm volatile("s_wait_asynccnt 0x0" ::: "memory");
            }
            __syncthreads();   // staged chunk visible to all waves
            const _Float16* Wst = stage + (mat & 1) * STGE;

#pragma unroll
            for (int nt = 0; nt < 4; ++nt) {
                v8f acc = {};
#pragma unroll
                for (int kk = 0; kk < 8; ++kk) {
                    v16h a = frag_A_rowmajor(xga, DIN, m0, kk * 32, lane);
                    v16h b = frag_B_nmajor(Wst, LDB, nt * 16, kk * 32, lane);
                    acc = __builtin_amdgcn_wmma_f32_16x16x32_f16(
                        false, a, false, b, (short)0, acc, false, false);
                }
                if (mat == 3) {                 // skip projection -> registers
                    const float bvv = bs[h * C + nt * 16 + (lane & 15)];
#pragma unroll
                    for (int i = 0; i < 8; ++i) acc[i] += bvv;
                    skipAcc[nt] = acc;
                } else if (mat == 2) {          // V -> LDS transposed [c][j]
                    const float bvv = bv[h * C + nt * 16 + (lane & 15)];
#pragma unroll
                    for (int i = 0; i < 8; ++i)
                        Vt[(nt * 16 + (lane & 15)) * LDV + m0 + i + half * 8] =
                            (_Float16)(acc[i] + bvv);
                } else {                        // Q/K -> LDS row-major
                    const float* bias = (mat == 0) ? bq : bk;
                    _Float16*    dst  = (mat == 0) ? Qh : Kh;
                    const float bvv = bias[h * C + nt * 16 + (lane & 15)];
#pragma unroll
                    for (int i = 0; i < 8; ++i)
                        dst[(m0 + i + half * 8) * LDQ + nt * 16 + (lane & 15)] =
                            (_Float16)(acc[i] + bvv);
                }
            }
            __syncthreads();   // chunk consumed; safe to overwrite next iter
        }

        // ---- scores = Q Kt for this wave's row tile (8 j-tiles) ------------
        v8f sc[8];
#pragma unroll
        for (int jt = 0; jt < 8; ++jt) {
            v8f acc = {};
#pragma unroll
            for (int cc = 0; cc < 2; ++cc) {
                v16h a = frag_A_rowmajor(Qh, LDQ, m0, cc * 32, lane);
                v16h b = frag_B_nmajor(Kh, LDQ, jt * 16, cc * 32, lane);
                acc = __builtin_amdgcn_wmma_f32_16x16x32_f16(
                    false, a, false, b, (short)0, acc, false, false);
            }
            sc[jt] = acc;
        }

        // ---- softmax over j (row = D-VGPR index, 16-lane half) -------------
        _Float16* myslab = slab + w * 16 * LDSLAB;
#pragma unroll
        for (int i = 0; i < 8; ++i) {
            float mval = -3.4e38f;
#pragma unroll
            for (int jt = 0; jt < 8; ++jt) mval = fmaxf(mval, sc[jt][i]);
#pragma unroll
            for (int off = 1; off < 16; off <<= 1)
                mval = fmaxf(mval, __shfl_xor(mval, off, 32));
            float ssum = 0.f;
#pragma unroll
            for (int jt = 0; jt < 8; ++jt) {
                float e = __expf(scale * (sc[jt][i] - mval));
                sc[jt][i] = e;
                ssum += e;
            }
#pragma unroll
            for (int off = 1; off < 16; off <<= 1)
                ssum += __shfl_xor(ssum, off, 32);
            const float rinv = 1.0f / ssum;
            const int   row  = i + half * 8;
#pragma unroll
            for (int jt = 0; jt < 8; ++jt)
                myslab[row * LDSLAB + jt * 16 + (lane & 15)] =
                    (_Float16)(sc[jt][i] * rinv);
        }
        // same-wave LDS write->read is in order; slab is wave-private.

        // ---- out = attn @ V + skip, then max over rows ---------------------
#pragma unroll
        for (int ct = 0; ct < 4; ++ct) {
            v8f acc = skipAcc[ct];
#pragma unroll
            for (int kk = 0; kk < 4; ++kk) {
                v16h a = frag_A_rowmajor(myslab, LDSLAB, 0, kk * 32, lane);
                v16h b = frag_B_nmajor(Vt, LDV, ct * 16, kk * 32, lane);
                acc = __builtin_amdgcn_wmma_f32_16x16x32_f16(
                    false, a, false, b, (short)0, acc, false, false);
            }
            float pm = acc[0];
#pragma unroll
            for (int i = 1; i < 8; ++i) pm = fmaxf(pm, acc[i]);
            pm = fmaxf(pm, __shfl_xor(pm, 16, 32));   // combine row halves
            if (lane < 16) pool[w * C + ct * 16 + lane] = pm;
        }
        __syncthreads();
        if (tid < C) {
            float m = pool[tid];
#pragma unroll
            for (int ww = 1; ww < 8; ++ww) m = fmaxf(m, pool[ww * C + tid]);
            out[(size_t)g * HC + h * C + tid] = m;
        }
        __syncthreads();   // protect Qh/Kh/Vt/pool reuse next head
    }
}

// ---------------------------------------------------------------------------
extern "C" void kernel_launch(void* const* d_in, const int* in_sizes, int n_in,
                              void* d_out, int out_size, void* d_ws, size_t ws_size,
                              hipStream_t stream)
{
    const float* x  = (const float*)d_in[0];
    // d_in[1] = subgbatch (contiguous equal groups; unused)
    const float* Wq = (const float*)d_in[2];
    const float* bq = (const float*)d_in[3];
    const float* Wk = (const float*)d_in[4];
    const float* bk = (const float*)d_in[5];
    const float* Wv = (const float*)d_in[6];
    const float* bv = (const float*)d_in[7];
    const float* Ws = (const float*)d_in[8];
    const float* bs = (const float*)d_in[9];

    // workspace: WT f16 [4][256][256] (512 KB) + x16 f16 [65536][256] (32 MB)
    _Float16* WT  = (_Float16*)d_ws;
    _Float16* x16 = WT + 4 * HC * DIN;

    prep_weights<<<(4 * HC * DIN) / 256, 256, 0, stream>>>(Wq, Wk, Wv, Ws, WT);
    prep_x<<<(65536 * DIN) / (4 * 256), 256, 0, stream>>>(x, x16);
    atten_pool_kernel<<<G, 256, 0, stream>>>(x16, WT, bq, bk, bv, bs,
                                             (float*)d_out);
}
